// DGATHead_48747878809859
// MI455X (gfx1250) — matrix operand order repaired and checked by reading, not compiled, not run on hardware
//
#include <hip/hip_runtime.h>
#include <hip/hip_bf16.h>

// GAT head, fused flash-attention style for MI455X (gfx1250, wave32).
//
// Roofline: adj (268 MB int32) read once = ~11.5 us @ 23.3 TB/s -> memory
// bound target. The 8192x8192 attention matrix never touches HBM: p tiles are
// built in-register and fed as the WMMA A-matrix of p@h
// (v_wmma_f32_16x16x32_f16). hT (64x8192 f16 = 1 MB) stays L2-resident.
//
// Softmax math, exactly refactored to kill hot-loop transcendentals:
//   exp(lrelu(s_i+d_j) - mr_i) = max(ua_i*va_j, ub_i*vb_j), all factors <= 1,
// with mr_i = lrelu(s_i + max_j d_j) an upper bound on the row max (lrelu is
// monotone) -> no online rescaling, partial j-ranges combine by addition.
// Row sums (softmax denominators) come from a 5th WMMA against all-ones B.

typedef _Float16 v16h __attribute__((ext_vector_type(16)));
typedef _Float16 v8h  __attribute__((ext_vector_type(8)));
typedef float    v8f  __attribute__((ext_vector_type(8)));

#define N_NODES 8192
#define D_IN    256
#define D_OUT   64
#define NEG_SLOPE 0.2f

__device__ __forceinline__ float lrelu(float x) { return fmaxf(x, NEG_SLOPE * x); }

__device__ __forceinline__ v16h cat8(v8h lo, v8h hi) {
    return __builtin_shufflevector(lo, hi, 0,1,2,3,4,5,6,7,8,9,10,11,12,13,14,15);
}

__device__ __forceinline__ float decode_max(unsigned u) {
    return (u & 0x80000000u) ? __uint_as_float(u ^ 0x80000000u)
                             : __uint_as_float(~u);
}

// p = adj ? exp(lrelu(s+d) - mr) : 0, via separable factors (no exp here)
__device__ __forceinline__ float pfun(int q, float vaj, float vbj,
                                      float ua, float ub) {
    float p = fmaxf(ua * vaj, ub * vbj);
    return (q > 0) ? p : 0.0f;
}

// ---------------------------------------------------------------- k0: init
__global__ void k_init(unsigned* encmax) {
    if (threadIdx.x == 0 && blockIdx.x == 0) *encmax = 0u;  // encoded -inf
}

// ------------------------------------------------- k1: h = X@W, store hT f16
__global__ void __launch_bounds__(128)
k_hproj(const float* __restrict__ X, const float* __restrict__ W,
        _Float16* __restrict__ hT)
{
    __shared__ _Float16 WT[D_OUT][D_IN];            // 32 KB, [d][k]
    for (int idx = threadIdx.x; idx < D_IN * D_OUT; idx += blockDim.x) {
        int k = idx >> 6, d = idx & 63;             // W row-major [k][d]
        WT[d][k] = (_Float16)W[idx];
    }
    __syncthreads();

    const int lane = threadIdx.x & 31;
    const int wv   = threadIdx.x >> 5;
    const int hi   = lane >> 4;                     // lane-half
    const int nn   = lane & 15;                     // row (A) / col (B,C)
    const int i0   = (blockIdx.x * 4 + wv) * 16;

    v8f acc[4] = {};
    const float* xr = X + (size_t)(i0 + nn) * D_IN;

    for (int k0 = 0; k0 < D_IN; k0 += 32) {
        const int kbA = k0 + hi * 8;                // A-layout K runs
        float4 x0 = *(const float4*)(xr + kbA);
        float4 x1 = *(const float4*)(xr + kbA + 4);
        float4 x2 = *(const float4*)(xr + kbA + 16);
        float4 x3 = *(const float4*)(xr + kbA + 20);
        v16h A;
        A[0]=(_Float16)x0.x;  A[1]=(_Float16)x0.y;  A[2]=(_Float16)x0.z;  A[3]=(_Float16)x0.w;
        A[4]=(_Float16)x1.x;  A[5]=(_Float16)x1.y;  A[6]=(_Float16)x1.z;  A[7]=(_Float16)x1.w;
        A[8]=(_Float16)x2.x;  A[9]=(_Float16)x2.y;  A[10]=(_Float16)x2.z; A[11]=(_Float16)x2.w;
        A[12]=(_Float16)x3.x; A[13]=(_Float16)x3.y; A[14]=(_Float16)x3.z; A[15]=(_Float16)x3.w;

        const int kbB = k0 + hi * 8;                // B-layout K runs (match A)
        #pragma unroll
        for (int b = 0; b < 4; ++b) {
            const _Float16* wp = &WT[b * 16 + nn][kbB];
            v16h B = cat8(*(const v8h*)wp, *(const v8h*)(wp + 16));
            acc[b] = __builtin_amdgcn_wmma_f32_16x16x32_f16(
                false, A, false, B, (short)0, acc[b], false, false);
        }
    }

    #pragma unroll
    for (int b = 0; b < 4; ++b) {
        v8h o;
        #pragma unroll
        for (int r = 0; r < 8; ++r) o[r] = (_Float16)acc[b][r];
        *(v8h*)(hT + (size_t)(b * 16 + nn) * N_NODES + i0 + 8 * hi) = o;
    }
}

// --------------------------------- k2: src/dst vectors + global max(dst)
__global__ void __launch_bounds__(256)
k_attvec(const _Float16* __restrict__ hT, const float* __restrict__ a,
         float* __restrict__ src, float* __restrict__ dst,
         unsigned* __restrict__ encmax)
{
    const int i = blockIdx.x * blockDim.x + threadIdx.x;
    float s0 = 0.f, s1 = 0.f;
    #pragma unroll
    for (int d = 0; d < D_OUT; ++d) {
        float hv = (float)hT[(size_t)d * N_NODES + i];
        s0 = fmaf(hv, a[d], s0);
        s1 = fmaf(hv, a[D_OUT + d], s1);
    }
    src[i] = s0; dst[i] = s1;

    float m = s1;
    #pragma unroll
    for (int off = 16; off > 0; off >>= 1) m = fmaxf(m, __shfl_xor(m, off, 32));
    if ((threadIdx.x & 31) == 0) {
        unsigned u = __float_as_uint(m);            // monotone float->uint encode
        u = (u & 0x80000000u) ? ~u : (u | 0x80000000u);
        atomicMax(encmax, u);
    }
}

// ---------------------- k2b: per-column exp factors (both <= 1, no overflow)
__global__ void __launch_bounds__(256)
k_vexp(const float* __restrict__ dst, const unsigned* __restrict__ encmax,
       float* __restrict__ va, float* __restrict__ vb)
{
    const int i = blockIdx.x * blockDim.x + threadIdx.x;
    const float dmax = decode_max(*encmax);
    const float t = dst[i] - dmax;                  // <= 0
    va[i] = __expf(t);
    vb[i] = __expf(NEG_SLOPE * t);
}

// ---------------- k3: fused masked-softmax + att@h + elu (flash style)
// One WG (4 waves) per 16-row tile; wave w owns j in [w*2048, (w+1)*2048).
// Per 32-col step: stream adj, build p in-register (2 mul + max + select,
// zero transcendentals) as the WMMA A-matrix, 4 WMMAs for p@h + 1 WMMA vs
// all-ones B for the denominator. Partials summed through LDS at the end.
__global__ void __launch_bounds__(128)
k_attn(const int* __restrict__ adj, const _Float16* __restrict__ hT,
       const float* __restrict__ src,
       const float* __restrict__ va, const float* __restrict__ vb,
       const unsigned* __restrict__ encmax, float* __restrict__ out)
{
    __shared__ float accbuf[4][16][64];   // 16 KB
    __shared__ float lbuf[4][16];

    const int lane = threadIdx.x & 31;
    const int wv   = threadIdx.x >> 5;    // j-quarter owned by this wave
    const int hi   = lane >> 4;
    const int nn   = lane & 15;
    const int i0   = blockIdx.x * 16;     // one row-tile per WG

    const float dmax = decode_max(*encmax);
    const float s  = src[i0 + nn];
    const float z  = s + dmax;
    const float mr = lrelu(z);                    // >= max_j e_ij
    const float ua = __expf(z - mr);              // <= 1
    const float ub = __expf(NEG_SLOPE * z - mr);  // <= 1

    v16h Bones;
    #pragma unroll
    for (int v = 0; v < 16; ++v) Bones[v] = (_Float16)1.0f;

    v8f acc[4] = {};
    v8f accl   = {};                              // p @ ones -> row sums
    const int* arow = adj + (size_t)(i0 + nn) * N_NODES;

    const int jlo = wv * (N_NODES / 4);
    const int jup = jlo + (N_NODES / 4);

    #pragma unroll 2
    for (int j0 = jlo; j0 < jup; j0 += 32) {
        const int jbA = j0 + hi * 8;              // A-layout K runs
        // prefetch the adj line this row needs 4 iterations from now
        __builtin_prefetch((const void*)(arow + jbA + 128), 0, 1);

        int4 q0 = *(const int4*)(arow + jbA);
        int4 q1 = *(const int4*)(arow + jbA + 4);
        int4 q2 = *(const int4*)(arow + jbA + 16);
        int4 q3 = *(const int4*)(arow + jbA + 20);
        float4 a0 = *(const float4*)(va + jbA);
        float4 a1 = *(const float4*)(va + jbA + 4);
        float4 a2 = *(const float4*)(va + jbA + 16);
        float4 a3 = *(const float4*)(va + jbA + 20);
        float4 b0 = *(const float4*)(vb + jbA);
        float4 b1 = *(const float4*)(vb + jbA + 4);
        float4 b2 = *(const float4*)(vb + jbA + 16);
        float4 b3 = *(const float4*)(vb + jbA + 20);

        v16h A;
        A[0] =(_Float16)pfun(q0.x,a0.x,b0.x,ua,ub); A[1] =(_Float16)pfun(q0.y,a0.y,b0.y,ua,ub);
        A[2] =(_Float16)pfun(q0.z,a0.z,b0.z,ua,ub); A[3] =(_Float16)pfun(q0.w,a0.w,b0.w,ua,ub);
        A[4] =(_Float16)pfun(q1.x,a1.x,b1.x,ua,ub); A[5] =(_Float16)pfun(q1.y,a1.y,b1.y,ua,ub);
        A[6] =(_Float16)pfun(q1.z,a1.z,b1.z,ua,ub); A[7] =(_Float16)pfun(q1.w,a1.w,b1.w,ua,ub);
        A[8] =(_Float16)pfun(q2.x,a2.x,b2.x,ua,ub); A[9] =(_Float16)pfun(q2.y,a2.y,b2.y,ua,ub);
        A[10]=(_Float16)pfun(q2.z,a2.z,b2.z,ua,ub); A[11]=(_Float16)pfun(q2.w,a2.w,b2.w,ua,ub);
        A[12]=(_Float16)pfun(q3.x,a3.x,b3.x,ua,ub); A[13]=(_Float16)pfun(q3.y,a3.y,b3.y,ua,ub);
        A[14]=(_Float16)pfun(q3.z,a3.z,b3.z,ua,ub); A[15]=(_Float16)pfun(q3.w,a3.w,b3.w,ua,ub);

        const int jbB = j0 + hi * 8;              // B-layout K runs (match A)
        #pragma unroll
        for (int b = 0; b < 4; ++b) {
            const _Float16* hp = hT + (size_t)(b * 16 + nn) * N_NODES + jbB;
            v16h B = cat8(*(const v8h*)hp, *(const v8h*)(hp + 16));
            acc[b] = __builtin_amdgcn_wmma_f32_16x16x32_f16(
                false, A, false, B, (short)0, acc[b], false, false);
        }
        accl = __builtin_amdgcn_wmma_f32_16x16x32_f16(
            false, A, false, Bones, (short)0, accl, false, false);
    }

    // publish this wave's partial tile + row sums
    #pragma unroll
    for (int b = 0; b < 4; ++b)
        #pragma unroll
        for (int r = 0; r < 8; ++r)
            accbuf[wv][r + 8 * hi][b * 16 + nn] = acc[b][r];
    if (nn == 0) {
        #pragma unroll
        for (int r = 0; r < 8; ++r) lbuf[wv][r + 8 * hi] = accl[r];
    }
    __syncthreads();

    // combine 4 partials, normalize, elu, store
    for (int idx = threadIdx.x; idx < 16 * 64; idx += 128) {
        const int row = idx >> 6, d = idx & 63;
        float sa = accbuf[0][row][d] + accbuf[1][row][d]
                 + accbuf[2][row][d] + accbuf[3][row][d];
        float l  = lbuf[0][row] + lbuf[1][row] + lbuf[2][row] + lbuf[3][row];
        float v  = sa / l;
        v = (v > 0.f) ? v : (__expf(v) - 1.0f);   // elu
        out[(size_t)(i0 + row) * D_OUT + d] = v;
    }
}

// ----------------------------------------------------------------- launch
extern "C" void kernel_launch(void* const* d_in, const int* in_sizes, int n_in,
                              void* d_out, int out_size, void* d_ws, size_t ws_size,
                              hipStream_t stream)
{
    const float* X   = (const float*)d_in[0];   // [1,8192,256] f32
    const int*   adj = (const int*)d_in[1];     // [8192,8192] i32
    const float* W   = (const float*)d_in[2];   // [256,64] f32
    const float* a   = (const float*)d_in[3];   // [128,1] f32
    float* out = (float*)d_out;                 // [8192,64] f32

    char* ws = (char*)d_ws;
    _Float16* hT    = (_Float16*)ws;                           // 1 MB
    float*    src   = (float*)(ws + (1u << 20));               // 32 KB
    float*    dst   = (float*)(ws + (1u << 20) + 32 * 1024);   // 32 KB
    float*    va    = (float*)(ws + (1u << 20) + 64 * 1024);   // 32 KB
    float*    vb    = (float*)(ws + (1u << 20) + 96 * 1024);   // 32 KB
    unsigned* encmx = (unsigned*)(ws + (1u << 20) + 128 * 1024);

    k_init  <<<1,   1,   0, stream>>>(encmx);
    k_hproj <<<128, 128, 0, stream>>>(X, W, hT);   // 512 row-tiles, 4 waves/blk
    k_attvec<<<32,  256, 0, stream>>>(hT, a, src, dst, encmx);
    k_vexp  <<<32,  256, 0, stream>>>(dst, encmx, va, vb);
    k_attn  <<<512, 128, 0, stream>>>(adj, hT, src, va, vb, encmx, out);
}
